// NNAKF_21131239097135
// MI455X (gfx1250) — matrix operand order repaired and checked by reading, not compiled, not run on hardware
//
#include <hip/hip_runtime.h>

// NNAKF: neural-network-adapted Kalman filter, sequential scan over T=512.
// Per-workgroup (16 batches) per-step LSTM recurrence h@W_hh^T done with
// v_wmma_f32_16x16x32_f16; W_hh (and the W_fc head) live in VGPRs as f16
// B-fragments for the entire time loop; h is exchanged via a 4KB LDS buffer.
// Activations use the CDNA5 hardware V_TANH_F32 transcendental when available.

typedef __attribute__((ext_vector_type(16))) _Float16 v16h;
typedef __attribute__((ext_vector_type(8)))  float    v8f;

#define T_STEPS 512
#define HID     128
#define NAD     10
#define DT_     0.1f
#define Q0V     0.01f      // 0.1^2
#define RV      0.25f      // 0.5^2

#if defined(__HIP_DEVICE_COMPILE__) && __has_builtin(__builtin_amdgcn_tanhf)
__device__ __forceinline__ float fast_tanh(float x) { return __builtin_amdgcn_tanhf(x); }
#else
// branch-free fallback: tanh(x) = 1 - 2/(e^{2x}+1), e^{2x} via raw v_exp_f32
__device__ __forceinline__ float fast_tanh(float x) {
  float e = __builtin_amdgcn_exp2f(x * 2.88539008178f); // 2*log2(e)
  return 1.0f - 2.0f / (e + 1.0f);
}
#endif
__device__ __forceinline__ float fast_sigm(float x) {
  return 0.5f * fast_tanh(0.5f * x) + 0.5f;
}

__global__ __launch_bounds__(256, 1)
void nnakf_scan_kernel(const float* __restrict__ meas,   // [B,T,2]
                       const float* __restrict__ Qt,     // [10,4,4]
                       const float* __restrict__ Wih,    // [512,2]
                       const float* __restrict__ Whh,    // [512,128]
                       const float* __restrict__ bih,    // [512]
                       const float* __restrict__ bhh,    // [512]
                       const float* __restrict__ Wfc,    // [10,128]
                       const float* __restrict__ bfc,    // [10]
                       float* __restrict__ out)          // [B,T,4]
{
  __shared__ _Float16 hbuf[16 * HID];       // h (f16), A-matrix source, 4 KB
  __shared__ float    s_in[16 * 2];         // normalized innovation In
  __shared__ float    s_sig[16 * NAD];      // sigma head output
  __shared__ float    s_qt[NAD * 16];       // Q_tilde flattened 4x4

  const int tid  = threadIdx.x;
  const int lane = tid & 31;
  const int w    = tid >> 5;        // wave 0..7
  const int p    = lane >> 4;       // half-wave 0/1
  const int nloc = lane & 15;
  const int kg   = w * 16 + nloc;   // this lane's hidden-column 0..127

  // ---------------- init: loop-invariant data ----------------
  for (int i = tid; i < 16 * HID; i += 256) hbuf[i] = (_Float16)0.0f;
  for (int i = tid; i < NAD * 16; i += 256) s_qt[i] = Qt[i];

  // per-lane gate-column constants: columns n = g*128 + kg, g in {i,f,g,o}
  float biasg[4], wih0[4], wih1[4];
#pragma unroll
  for (int g = 0; g < 4; ++g) {
    const int n = g * HID + kg;
    biasg[g] = bih[n] + bhh[n];
    wih0[g]  = Wih[2 * n + 0];
    wih1[g]  = Wih[2 * n + 1];
  }

  // W_hh -> persistent f16 B-fragments (B[k][n] = W_hh[n][k]).
  // B k order here (contiguous, as A): lane<16 -> col N=lane, K=0..15; lane>=16 -> col N=lane-16, K=16..31.
  v16h bfrag[4][4];
#pragma unroll
  for (int g = 0; g < 4; ++g) {
    const int n = g * HID + kg;
#pragma unroll
    for (int k = 0; k < 4; ++k) {
      const float* src = Whh + n * HID + k * 32 + p * 16;
#pragma unroll
      for (int j = 0; j < 16; ++j) bfrag[g][k][j] = (_Float16)src[j];
    }
  }

  // W_fc^T padded to N=16 -> persistent B-fragments for the sigma head
  // (only wave 0 uses them; B[k][n] = W_fc[n][k], n>=10 -> 0)
  v16h bfragfc[4];
  float bfc_l = (nloc < NAD) ? bfc[nloc] : 0.0f;
#pragma unroll
  for (int k = 0; k < 4; ++k) {
#pragma unroll
    for (int j = 0; j < 16; ++j) {
      float v = (nloc < NAD) ? Wfc[nloc * HID + k * 32 + p * 16 + j] : 0.0f;
      bfragfc[k][j] = (_Float16)v;
    }
  }

  // ---------------- per-batch Kalman state (thread b<16 owns batch b) -------
  float X[4]    = {0.f, 0.f, 0.f, 0.f};
  float P[4][4] = {{1,0,0,0},{0,1,0,0},{0,0,1,0},{0,0,0,1}};
  float Pp[4][4];
  float xp[4];
  float iv0 = 0.f, iv1 = 0.f;
  float cst[8] = {0,0,0,0,0,0,0,0};   // cell state slice: 8 batches (rows 8p..8p+7) x this k-col

  const float* zptr = meas + (size_t)(blockIdx.x * 16 + (tid & 15)) * (2 * T_STEPS);
  float zc0 = 0.f, zc1 = 0.f;
  if (tid < 16) { zc0 = zptr[0]; zc1 = zptr[1]; }

  __syncthreads();

  for (int t = 0; t < T_STEPS; ++t) {
    // ---- (1) Kalman predict + normalized innovation (threads 0..15) ----
    if (tid < 16) {
      float FP[4][4];
#pragma unroll
      for (int j = 0; j < 4; ++j) {
        FP[0][j] = P[0][j] + DT_ * P[2][j];
        FP[1][j] = P[1][j] + DT_ * P[3][j];
        FP[2][j] = P[2][j];
        FP[3][j] = P[3][j];
      }
#pragma unroll
      for (int i = 0; i < 4; ++i) {
        Pp[i][0] = FP[i][0] + DT_ * FP[i][2];
        Pp[i][1] = FP[i][1] + DT_ * FP[i][3];
        Pp[i][2] = FP[i][2];
        Pp[i][3] = FP[i][3];
      }
#pragma unroll
      for (int i = 0; i < 4; ++i) Pp[i][i] += Q0V;
      xp[0] = X[0] + DT_ * X[2];
      xp[1] = X[1] + DT_ * X[3];
      xp[2] = X[2];
      xp[3] = X[3];
      iv0 = zc0 - xp[0];
      iv1 = zc1 - xp[1];
      s_in[2 * tid + 0] = iv0 * iv0 / (Pp[0][0] + RV);
      s_in[2 * tid + 1] = iv1 * iv1 / (Pp[1][1] + RV);
      if (t + 1 < T_STEPS) { zc0 = zptr[2 * (t + 1)]; zc1 = zptr[2 * (t + 1) + 1]; }
    }
    __syncthreads();

    // ---- (2) A-fragments of h from LDS (16-bit A 16x32 layout) ----
    v16h afr[4];
#pragma unroll
    for (int k = 0; k < 4; ++k) {
      const _Float16* s = &hbuf[nloc * HID + k * 32 + p * 16];
      union { float4 f[2]; v16h v; } u;
      u.f[0] = *(const float4*)(s);        // K = k*32 + p*16 .. +7
      u.f[1] = *(const float4*)(s + 8);    // K = k*32 + p*16 + 8 .. +15
      afr[k] = u.v;
    }
    __syncthreads();   // all h reads done before h is overwritten below

    // ---- (3) gates = h @ W_hh^T : 16 WMMAs, W_hh resident in VGPRs ----
    v8f acc[4];
#pragma unroll
    for (int g = 0; g < 4; ++g) {
      v8f z = {0.f, 0.f, 0.f, 0.f, 0.f, 0.f, 0.f, 0.f};
      acc[g] = z;
#pragma unroll
      for (int k = 0; k < 4; ++k) {
        acc[g] = __builtin_amdgcn_wmma_f32_16x16x32_f16(
            false, afr[k], false, bfrag[g][k], (short)0, acc[g], false, false);
      }
    }

    // ---- (4) LSTM pointwise (i/f/g/o element-aligned in the 4 accs) ----
#pragma unroll
    for (int r = 0; r < 8; ++r) {
      const int m = p * 8 + r;           // batch row per C/D layout
      const float i0 = s_in[2 * m + 0];
      const float i1 = s_in[2 * m + 1];
      const float gi = acc[0][r] + biasg[0] + i0 * wih0[0] + i1 * wih1[0];
      const float gf = acc[1][r] + biasg[1] + i0 * wih0[1] + i1 * wih1[1];
      const float gg = acc[2][r] + biasg[2] + i0 * wih0[2] + i1 * wih1[2];
      const float go = acc[3][r] + biasg[3] + i0 * wih0[3] + i1 * wih1[3];
      const float cn = fast_sigm(gf) * cst[r] + fast_sigm(gi) * fast_tanh(gg);
      cst[r] = cn;
      const float hn = fast_sigm(go) * fast_tanh(cn);
      hbuf[m * HID + kg] = (_Float16)hn;
    }
    __syncthreads();

    // ---- (5) sigma = sigmoid(h @ W_fc^T + b_fc): 4 more WMMAs on wave 0 ----
    if (w == 0) {   // wave-uniform: EXEC all-ones within the executing wave
      v16h a2[4];
#pragma unroll
      for (int k = 0; k < 4; ++k) {
        const _Float16* s = &hbuf[nloc * HID + k * 32 + p * 16];
        union { float4 f[2]; v16h v; } u;
        u.f[0] = *(const float4*)(s);
        u.f[1] = *(const float4*)(s + 8);
        a2[k] = u.v;
      }
      v8f accs = {0.f, 0.f, 0.f, 0.f, 0.f, 0.f, 0.f, 0.f};
#pragma unroll
      for (int k = 0; k < 4; ++k) {
        accs = __builtin_amdgcn_wmma_f32_16x16x32_f16(
            false, a2[k], false, bfragfc[k], (short)0, accs, false, false);
      }
      if (nloc < NAD) {
#pragma unroll
        for (int r = 0; r < 8; ++r)
          s_sig[(p * 8 + r) * NAD + nloc] = fast_sigm(accs[r] + bfc_l);
      }
    }
    __syncthreads();

    // ---- (6) Q-adapt + Kalman update + output (threads 0..15) ----
    if (tid < 16) {
      float* Ppf = &Pp[0][0];
#pragma unroll
      for (int n = 0; n < NAD; ++n) {
        const float sg = s_sig[tid * NAD + n];
#pragma unroll
        for (int e = 0; e < 16; ++e) Ppf[e] += sg * s_qt[n * 16 + e];
      }
      const float S00 = Pp[0][0] + RV, S01 = Pp[0][1];
      const float S10 = Pp[1][0],      S11 = Pp[1][1] + RV;
      const float idet = 1.0f / (S00 * S11 - S01 * S10);
      const float i00 =  S11 * idet, i01 = -S01 * idet;
      const float i10 = -S10 * idet, i11 =  S00 * idet;
      float K0[4], K1[4];
#pragma unroll
      for (int i = 0; i < 4; ++i) {
        K0[i] = Pp[i][0] * i00 + Pp[i][1] * i10;
        K1[i] = Pp[i][0] * i01 + Pp[i][1] * i11;
      }
      float nx[4];
#pragma unroll
      for (int i = 0; i < 4; ++i) nx[i] = xp[i] + K0[i] * iv0 + K1[i] * iv1;
#pragma unroll
      for (int i = 0; i < 4; ++i) {
#pragma unroll
        for (int j = 0; j < 4; ++j)
          P[i][j] = Pp[i][j] - K0[i] * Pp[0][j] - K1[i] * Pp[1][j];
        X[i] = nx[i];
      }
      float4 o;
      o.x = nx[0]; o.y = nx[1]; o.z = nx[2]; o.w = nx[3];
      *(float4*)(out + ((size_t)(blockIdx.x * 16 + tid) * T_STEPS + t) * 4) = o;
    }
    // no barrier needed here: next-iter writers of s_in are the same threads,
    // and all cross-thread readers are synced by the barriers above.
  }
}

extern "C" void kernel_launch(void* const* d_in, const int* in_sizes, int n_in,
                              void* d_out, int out_size, void* d_ws, size_t ws_size,
                              hipStream_t stream) {
  const float* meas = (const float*)d_in[0];
  const float* Qt   = (const float*)d_in[1];
  const float* Wih  = (const float*)d_in[2];
  const float* Whh  = (const float*)d_in[3];
  const float* bih  = (const float*)d_in[4];
  const float* bhh  = (const float*)d_in[5];
  const float* Wfc  = (const float*)d_in[6];
  const float* bfc  = (const float*)d_in[7];
  float* out = (float*)d_out;

  const int B = in_sizes[0] / (2 * T_STEPS);   // 1024
  const int blocks = B / 16;                   // 64 workgroups, 16 batches each
  hipLaunchKernelGGL(nnakf_scan_kernel, dim3(blocks), dim3(256), 0, stream,
                     meas, Qt, Wih, Whh, bih, bhh, Wfc, bfc, out);
}
